// ItemQueryAttention_75728863363479
// MI455X (gfx1250) — matrix operand order repaired and checked by reading, not compiled, not run on hardware
//
#include <hip/hip_runtime.h>
#include <hip/hip_bf16.h>

// Problem dims (fixed by reference)
#define QQ 32
#define SS 25
#define LL 64
#define FF 256
#define HH 64

#if __has_builtin(__builtin_amdgcn_tanhf)
#define TANHF(x) __builtin_amdgcn_tanhf(x)
#elif __has_builtin(__builtin_amdgcn_tanh_f32)
#define TANHF(x) __builtin_amdgcn_tanh_f32(x)
#else
#define TANHF(x) tanhf(x)
#endif

typedef __attribute__((ext_vector_type(16))) __bf16 v16bf;
typedef __attribute__((ext_vector_type(8)))  float v8f;

union Frag {
    v16bf v;
    unsigned int u[8];
    unsigned short h[16];
};

__device__ __forceinline__ unsigned short f32_to_bf16(float f) {
    unsigned int u = __builtin_bit_cast(unsigned int, f);
    u += 0x7FFFu + ((u >> 16) & 1u);   // round-to-nearest-even
    return (unsigned short)(u >> 16);
}

// ---------------------------------------------------------------------------
// K0: hs (f32) -> hs_bf16
// ---------------------------------------------------------------------------
__global__ void k_hs_bf16(const float* __restrict__ hs,
                          unsigned short* __restrict__ hsb, int n) {
    int i = blockIdx.x * blockDim.x + threadIdx.x;
    if (i < n) hsb[i] = f32_to_bf16(hs[i]);
}

// ---------------------------------------------------------------------------
// K1: projections  Wq[q,l,h] = qs[q,l,:]·W[h,:] + b[h]   (same for Wh from hs)
// 4 rows per 256-thread block; W staged in LDS with +1 pad (bank-conflict free)
// ---------------------------------------------------------------------------
__global__ __launch_bounds__(256) void k_project(const float* __restrict__ qs,
                                                 const float* __restrict__ hs,
                                                 const float* __restrict__ W,
                                                 const float* __restrict__ b,
                                                 float* __restrict__ wq,
                                                 float* __restrict__ wh) {
    __shared__ float wlds[HH * (FF + 1)];
    __shared__ float xr[4][FF];
    __shared__ float blds[HH];

    for (int idx = threadIdx.x; idx < HH * FF; idx += 256) {
        int h = idx / FF, f = idx % FF;
        wlds[h * (FF + 1) + f] = W[idx];
    }
    if (threadIdx.x < HH) blds[threadIdx.x] = b[threadIdx.x];

    const int row0 = blockIdx.x * 4;
    for (int r = 0; r < 4; ++r) {
        int row = row0 + r;
        const float* src = (row < QQ * LL) ? (qs + (size_t)row * FF)
                                           : (hs + (size_t)(row - QQ * LL) * FF);
        for (int f = threadIdx.x; f < FF; f += 256) xr[r][f] = src[f];
    }
    __syncthreads();

    const int r = threadIdx.x >> 6;      // 0..3
    const int h = threadIdx.x & 63;      // 0..63
    const int row = row0 + r;
    float acc = blds[h];
    const float* wrow = &wlds[h * (FF + 1)];
    const float* x = xr[r];
#pragma unroll 8
    for (int f = 0; f < FF; ++f) acc += x[f] * wrow[f];

    if (row < QQ * LL) wq[(size_t)row * HH + h] = acc;
    else               wh[(size_t)(row - QQ * LL) * HH + h] = acc;
}

// ---------------------------------------------------------------------------
// K2: scores[i,j] = sum_h tanh(wq[i,h]*wh[j,h]); softmax over j; att -> bf16
// One block per (q,s). LDS tiles padded (stride 65) -> conflict-free.
// ---------------------------------------------------------------------------
__global__ __launch_bounds__(256) void k_scores(const float* __restrict__ wq,
                                                const float* __restrict__ wh,
                                                unsigned short* __restrict__ att) {
    const int q = blockIdx.x / SS;
    const int s = blockIdx.x % SS;

    __shared__ float wqs[LL][HH + 1];
    __shared__ float whs[LL][HH + 1];
    __shared__ float sc[LL][LL + 1];

    const float* wqsrc = wq + (size_t)q * LL * HH;
    const float* whsrc = wh + (size_t)s * LL * HH;
    for (int idx = threadIdx.x; idx < LL * HH; idx += 256) {
        int i = idx >> 6, h = idx & 63;
        wqs[i][h] = wqsrc[idx];
        whs[i][h] = whsrc[idx];
    }
    __syncthreads();

    // 4096 score entries, 16 per thread. i is wave-uniform -> wqs broadcast;
    // whs stride 65 dwords across lanes -> distinct banks.
#pragma unroll
    for (int k = 0; k < 16; ++k) {
        int idx = threadIdx.x + k * 256;
        int i = idx >> 6, j = idx & 63;
        float acc = 0.f;
#pragma unroll 8
        for (int h = 0; h < HH; ++h) acc += TANHF(wqs[i][h] * whs[j][h]);
        sc[i][j] = acc;
    }
    __syncthreads();

    if (threadIdx.x < LL) {
        const int i = threadIdx.x;
        float m = -3.0e38f;
        for (int j = 0; j < LL; ++j) m = fmaxf(m, sc[i][j]);
        float sum = 0.f;
        for (int j = 0; j < LL; ++j) {
            float e = __builtin_amdgcn_exp2f((sc[i][j] - m) * 1.4426950408889634f);
            sc[i][j] = e;
            sum += e;
        }
        float inv = 1.0f / sum;
        for (int j = 0; j < LL; ++j) sc[i][j] *= inv;
    }
    __syncthreads();

    unsigned short* dst = att + ((size_t)q * SS + s) * LL * LL;
#pragma unroll
    for (int k = 0; k < 16; ++k) {
        int idx = threadIdx.x + k * 256;
        dst[idx] = f32_to_bf16(sc[idx >> 6][idx & 63]);
    }
}

// ---------------------------------------------------------------------------
// K3: out[q,s] (64x256 f32) = att[q,s] (64x64 bf16) @ hs[s] (64x256 bf16)
// One block (8 waves) per (q,s). hs[s] transposed into LDS (f-major, +2 pad:
// row stride 33 dwords, coprime with 64 banks). v_wmma_f32_16x16x32_bf16,
// 2 K-steps, 64 D-tiles per block.
// ---------------------------------------------------------------------------
__global__ __launch_bounds__(256) void k_out(const unsigned short* __restrict__ att,
                                             const unsigned short* __restrict__ hsb,
                                             float* __restrict__ out) {
    const int q = blockIdx.x / SS;
    const int s = blockIdx.x % SS;

    __shared__ unsigned short hsT[FF][LL + 2];   // [f][j], 33792 bytes

    const unsigned short* hsrc = hsb + (size_t)s * LL * FF;
    for (int idx = threadIdx.x; idx < LL * FF; idx += 256) {
        int j = idx >> 8;          // / FF
        int f = idx & (FF - 1);
        hsT[f][j] = hsrc[idx];
    }
    __syncthreads();

    const int wave = threadIdx.x >> 5;
    const int lane = threadIdx.x & 31;
    const int m0 = (wave & 3) * 16;          // M tile (i)
    const int nbase = (wave >> 2) * 128;     // N half (f)
    const int r16 = lane & 15;
    const int half = lane >> 4;

    // A fragments (contiguous k order, same as B): lane row = m0+r16; element pairs at
    // K = 2v + 16*half; second fragment offset +32.
    const unsigned short* arow =
        att + (((size_t)q * SS + s) * LL + (m0 + r16)) * LL;
    Frag a0, a1;
#pragma unroll
    for (int v = 0; v < 8; ++v) {
        int kk = 2 * v + half * 16;
        a0.u[v] = *(const unsigned int*)(arow + kk);
        a1.u[v] = *(const unsigned int*)(arow + 32 + kk);
    }

    const size_t obase = (((size_t)q * SS + s) * LL + m0 + half * 8) * FF;

    for (int nt = 0; nt < 8; ++nt) {
        const int n0 = nbase + nt * 16;
        // B fragments (contiguous k order, same as A): lane col = n0+r16, K = 16*half + e
        const unsigned short* bp0 = &hsT[n0 + r16][half * 16];
        const unsigned short* bp1 = &hsT[n0 + r16][32 + half * 16];
        Frag b0, b1;
#pragma unroll
        for (int v = 0; v < 8; ++v) {
            b0.u[v] = *(const unsigned int*)(bp0 + 2 * v);
            b1.u[v] = *(const unsigned int*)(bp1 + 2 * v);
        }
        v8f c = {};
        c = __builtin_amdgcn_wmma_f32_16x16x32_bf16(false, a0.v, false, b0.v,
                                                    (short)0, c, false, false);
        c = __builtin_amdgcn_wmma_f32_16x16x32_bf16(false, a1.v, false, b1.v,
                                                    (short)0, c, false, false);
        // D layout: element r -> row m0 + 8*half + r, col n0 + r16
        float* op = out + obase + n0 + r16;
#pragma unroll
        for (int r = 0; r < 8; ++r) op[(size_t)r * FF] = c[r];
    }
}

// ---------------------------------------------------------------------------
extern "C" void kernel_launch(void* const* d_in, const int* in_sizes, int n_in,
                              void* d_out, int out_size, void* d_ws, size_t ws_size,
                              hipStream_t stream) {
    const float* qs = (const float*)d_in[0];   // [Q,L,F]
    const float* hs = (const float*)d_in[1];   // [S,L,F]
    const float* W  = (const float*)d_in[2];   // [H,F]
    const float* b  = (const float*)d_in[3];   // [H]
    float* out = (float*)d_out;                // [Q,S,L,F]

    char* ws = (char*)d_ws;
    // Workspace layout (bytes):
    //   wq  : QQ*LL*HH f32      = 524288
    //   wh  : SS*LL*HH f32      = 409600
    //   hsb : SS*LL*FF bf16     = 819200
    //   att : QQ*SS*LL*LL bf16  = 6553600      total ~8.3 MB
    float*          wq   = (float*)(ws + 0);
    float*          wh   = (float*)(ws + 524288);
    unsigned short* hsb  = (unsigned short*)(ws + 524288 + 409600);
    unsigned short* attb = (unsigned short*)(ws + 524288 + 409600 + 819200);

    const int n_hs = SS * LL * FF;                      // 409600
    k_hs_bf16<<<(n_hs + 255) / 256, 256, 0, stream>>>(hs, hsb, n_hs);

    const int n_rows = QQ * LL + SS * LL;               // 3648
    k_project<<<n_rows / 4, 256, 0, stream>>>(qs, hs, W, b, wq, wh);

    k_scores<<<QQ * SS, 256, 0, stream>>>(wq, wh, attb);

    k_out<<<QQ * SS, 256, 0, stream>>>(attb, hsb, out);
}
